// InterAttention_50517405336002
// MI455X (gfx1250) — matrix operand order repaired and checked by reading, not compiled, not run on hardware
//
#include <hip/hip_runtime.h>

// InterAttention for MI455X (gfx1250): all GEMMs via v_wmma_f32_16x16x32_bf16,
// dual flash-attention (softmax(QK')K + softmax(VK')K) streamed per 16-row strip,
// K blocks staged into LDS with async global->LDS copies (double buffered).

#define B_   4
#define L_   2048
#define D_   1024
#define H_   8
#define DK_  128
#define BH_  32

typedef __attribute__((ext_vector_type(16))) __bf16 v16bf;
typedef __attribute__((ext_vector_type(8)))  __bf16 v8bf;
typedef __attribute__((ext_vector_type(4)))  __bf16 v4bf;
typedef __attribute__((ext_vector_type(8)))  float  v8f;
typedef __attribute__((ext_vector_type(4)))  int    v4i;

#if __has_builtin(__builtin_amdgcn_global_load_async_to_lds_b128)
#define HAVE_ASYNC_LDS 1
typedef __attribute__((address_space(1))) v4i gas_v4i;  // global (AS1) int4
typedef __attribute__((address_space(3))) v4i las_v4i;  // LDS (AS3) int4
#else
#define HAVE_ASYNC_LDS 0
#endif

// one 16-byte async (or fallback sync) copy global -> LDS
static __device__ __forceinline__ void async_copy16(__bf16* lds, const __bf16* gsrc) {
#if HAVE_ASYNC_LDS
  __builtin_amdgcn_global_load_async_to_lds_b128((gas_v4i*)gsrc, (las_v4i*)lds, 0, 0);
#else
  *(v8bf*)lds = *(const v8bf*)gsrc;
#endif
}

static __device__ __forceinline__ void async_wait0() {
#if HAVE_ASYNC_LDS
#if __has_builtin(__builtin_amdgcn_s_wait_asynccnt)
  __builtin_amdgcn_s_wait_asynccnt(0);
#else
  asm volatile("s_wait_asynccnt 0x0" ::: "memory");
#endif
#endif
}

static __device__ __forceinline__ v8f wmma_bf16(v16bf a, v16bf b, v8f c) {
  // D = A(16x32 bf16) x B(32x16 bf16) + C(16x16 f32)
  return __builtin_amdgcn_wmma_f32_16x16x32_bf16(false, a, false, b, (short)0, c,
                                                 false, false);
}

// A-fragment (16x32, row-major source, K contiguous, row stride ld elements).
// lane<16 -> row=lane, K 0..7 & 16..23 ; lane>=16 -> row=lane-16, K 8..15 & 24..31.
static __device__ __forceinline__ v16bf load_a_frag(const __bf16* base, int ld) {
  const int lane = threadIdx.x & 31;
  const int half = lane >> 4;
  const int r    = lane & 15;
  const __bf16* p = base + (size_t)r * ld + half * 8;
  v8bf c0 = *(const v8bf*)(p);
  v8bf c1 = *(const v8bf*)(p + 16);
  v16bf f;
#pragma unroll
  for (int i = 0; i < 8; ++i) { f[i] = c0[i]; f[i + 8] = c1[i]; }
  return f;
}

// B-fragment (32x16). Memory holds the 16 B-columns as rows (K contiguous, stride ld).
// lane<16 -> col=lane, K 0..7 & 16..23 ; lane>=16 -> col=lane-16, K 8..15 & 24..31.
static __device__ __forceinline__ v16bf load_b_frag(const __bf16* base, int ld) {
  const int lane = threadIdx.x & 31;
  const int half = lane >> 4;
  const int n    = lane & 15;
  const __bf16* p = base + (size_t)n * ld + half * 8;
  v8bf c0 = *(const v8bf*)(p);
  v8bf c1 = *(const v8bf*)(p + 16);
  v16bf f;
#pragma unroll
  for (int i = 0; i < 8; ++i) { f[i] = c0[i]; f[i + 8] = c1[i]; }
  return f;
}

// ---------------- Kernel 0: f32 -> bf16 pack ----------------
__global__ void ia_cvt_bf16(const float* __restrict__ in, __bf16* __restrict__ out, int n4) {
  int i = blockIdx.x * blockDim.x + threadIdx.x;
  if (i < n4) {
    float4 v = ((const float4*)in)[i];
    v4bf o;
    o[0] = (__bf16)v.x; o[1] = (__bf16)v.y; o[2] = (__bf16)v.z; o[3] = (__bf16)v.w;
    ((v4bf*)out)[i] = o;
  }
}

// ---------------- Kernel 1: Yt[b][d][t] = sum_c W[d][c] * X[b][t][c] ----------------
__global__ void ia_proj(const __bf16* __restrict__ X, const __bf16* __restrict__ W,
                        __bf16* __restrict__ Yt) {
  int w = (blockIdx.x * blockDim.x + threadIdx.x) >> 5;
  int batch = w >> 11;
  int rem   = w & 2047;
  int d0 = (rem >> 6) * 32;
  int t0 = (rem & 63) * 32;
  const __bf16* Xb = X + (size_t)batch * L_ * D_;
  v8f acc[2][2] = {};
#pragma unroll 4
  for (int kk = 0; kk < D_; kk += 32) {
    v16bf a0 = load_a_frag(W + (size_t)d0 * D_ + kk, D_);
    v16bf a1 = load_a_frag(W + (size_t)(d0 + 16) * D_ + kk, D_);
    v16bf b0 = load_b_frag(Xb + (size_t)t0 * D_ + kk, D_);
    v16bf b1 = load_b_frag(Xb + (size_t)(t0 + 16) * D_ + kk, D_);
    acc[0][0] = wmma_bf16(a0, b0, acc[0][0]);
    acc[0][1] = wmma_bf16(a0, b1, acc[0][1]);
    acc[1][0] = wmma_bf16(a1, b0, acc[1][0]);
    acc[1][1] = wmma_bf16(a1, b1, acc[1][1]);
  }
  const int lane = threadIdx.x & 31, half = lane >> 4, nn = lane & 15;
  __bf16* Yb = Yt + (size_t)batch * D_ * L_;
#pragma unroll
  for (int i = 0; i < 2; ++i)
#pragma unroll
    for (int j = 0; j < 2; ++j)
#pragma unroll
      for (int v = 0; v < 8; ++v)
        Yb[(size_t)(d0 + i * 16 + v + 8 * half) * L_ + (t0 + j * 16 + nn)] =
            (__bf16)acc[i][j][v];
}

// ---------------- Kernel 2: KT2[bh][k][m] = Kt[bh][m][k] ----------------
__global__ void ia_transpose(const __bf16* __restrict__ Kt, __bf16* __restrict__ KT2) {
  __shared__ __bf16 tile[32][33];
  int bh = blockIdx.z;
  int m0 = blockIdx.y * 32;
  int k0 = blockIdx.x * 32;
  const __bf16* src = Kt + (size_t)bh * L_ * DK_;
  __bf16* dst = KT2 + (size_t)bh * DK_ * L_;
  int x = threadIdx.x & 31, y = threadIdx.x >> 5;
#pragma unroll
  for (int i = 0; i < 4; ++i)
    tile[y * 4 + i][x] = src[(size_t)(m0 + y * 4 + i) * DK_ + (k0 + x)];
  __syncthreads();
#pragma unroll
  for (int i = 0; i < 4; ++i)
    dst[(size_t)(k0 + y * 4 + i) * L_ + (m0 + x)] = tile[x][y * 4 + i];
}

// ---------------- Kernel 3: dual flash attention ----------------
static __device__ __forceinline__ void online_update(v8f s0, v8f s1,
                                                     float* mstat, float* sstat,
                                                     v8f* O, __bf16* pb) {
  const int lane = threadIdx.x & 31, half = lane >> 4, ln = lane & 15;
  const float sc = 0.08838834764831845f;  // dk^-0.5
#pragma unroll
  for (int v = 0; v < 8; ++v) {
    float e0 = s0[v] * sc, e1 = s1[v] * sc;
    float t = fmaxf(e0, e1);
    t = fmaxf(t, __shfl_xor(t, 1, 32));
    t = fmaxf(t, __shfl_xor(t, 2, 32));
    t = fmaxf(t, __shfl_xor(t, 4, 32));
    t = fmaxf(t, __shfl_xor(t, 8, 32));
    float mnew = fmaxf(mstat[v], t);
    float corr = __expf(mstat[v] - mnew);
    mstat[v] = mnew;
    float p0 = __expf(e0 - mnew);
    float p1 = __expf(e1 - mnew);
    float rs = p0 + p1;
    rs += __shfl_xor(rs, 1, 32);
    rs += __shfl_xor(rs, 2, 32);
    rs += __shfl_xor(rs, 4, 32);
    rs += __shfl_xor(rs, 8, 32);
    sstat[v] = sstat[v] * corr + rs;
#pragma unroll
    for (int j = 0; j < 8; ++j) O[j][v] *= corr;
    pb[(v + 8 * half) * 32 + ln]      = (__bf16)p0;
    pb[(v + 8 * half) * 32 + 16 + ln] = (__bf16)p1;
  }
}

// stage one 32-key block of K (row-major) and K^T (dk-major) into LDS
static __device__ __forceinline__ void stage_kblock(__bf16* kb, __bf16* k2b,
                                                    const __bf16* Kb, const __bf16* K2g,
                                                    int cb, int tid) {
  // Kblk: Kt rows cb..cb+31 are one contiguous 8 KB span
  const __bf16* gk = Kb + (size_t)cb * DK_;
  async_copy16(kb + tid * 8, gk + tid * 8);
  async_copy16(kb + 2048 + tid * 8, gk + 2048 + tid * 8);
  // K2blk: 128 dk-rows x 32 keys (64 B per row, stride L_)
  int r0 = tid >> 2, w0 = tid & 3;
  async_copy16(k2b + r0 * 32 + w0 * 8, K2g + (size_t)r0 * L_ + cb + w0 * 8);
  int c1 = tid + 256;
  int r1 = c1 >> 2, w1 = c1 & 3;
  async_copy16(k2b + r1 * 32 + w1 * 8, K2g + (size_t)r1 * L_ + cb + w1 * 8);
}

__global__ void ia_attn(const __bf16* __restrict__ Qt, const __bf16* __restrict__ Kt,
                        const __bf16* __restrict__ Vt, const __bf16* __restrict__ KT2,
                        __bf16* __restrict__ Hb) {
  __shared__ __align__(16) __bf16 kbuf[2][32 * 128];    // 16 KB (double-buffered K)
  __shared__ __align__(16) __bf16 k2buf[2][128 * 32];   // 16 KB (double-buffered K^T)
  __shared__ __align__(16) __bf16 pbuf[8][2][16 * 32];  // 16 KB per-wave P staging

  const int tid = threadIdx.x;
  const int wid = tid >> 5;
  const int bh  = blockIdx.x >> 4;                 // 8 strips/block, 128 strips per bh
  const int m0  = ((blockIdx.x & 15) * 8 + wid) * 16;
  const int lane = tid & 31, half = lane >> 4, ln = lane & 15;

  const __bf16* Qb = Qt + (size_t)bh * L_ * DK_;
  const __bf16* Kb = Kt + (size_t)bh * L_ * DK_;
  const __bf16* Vb = Vt + (size_t)bh * L_ * DK_;
  const __bf16* K2 = KT2 + (size_t)bh * DK_ * L_;

  stage_kblock(kbuf[0], k2buf[0], Kb, K2, 0, tid);   // prologue: block 0 in flight

  v16bf qf[4], vf[4];
#pragma unroll
  for (int ks = 0; ks < 4; ++ks) {
    qf[ks] = load_a_frag(Qb + (size_t)m0 * DK_ + ks * 32, DK_);
    vf[ks] = load_a_frag(Vb + (size_t)m0 * DK_ + ks * 32, DK_);
  }

  v8f O1[8] = {}, O2[8] = {};
  float m1[8], s1[8], m2[8], s2[8];
#pragma unroll
  for (int v = 0; v < 8; ++v) { m1[v] = -1e30f; m2[v] = -1e30f; s1[v] = 0.f; s2[v] = 0.f; }

  for (int ii = 0; ii < L_ / 32; ++ii) {
    async_wait0();        // own copies for buffer ii&1 done
    __syncthreads();      // everyone's copies done; prior reads of other buffer done
    if (ii + 1 < L_ / 32)
      stage_kblock(kbuf[(ii + 1) & 1], k2buf[(ii + 1) & 1], Kb, K2, (ii + 1) * 32, tid);

    const __bf16* kb  = kbuf[ii & 1];
    const __bf16* k2b = k2buf[ii & 1];

    v8f S1a = {}, S1b = {}, S2a = {}, S2b = {};
#pragma unroll
    for (int ks = 0; ks < 4; ++ks) {  // accumulate over dk=128 from LDS
      v16bf b0 = load_b_frag(kb + ks * 32, DK_);
      v16bf b1 = load_b_frag(kb + 16 * DK_ + ks * 32, DK_);
      S1a = wmma_bf16(qf[ks], b0, S1a);
      S1b = wmma_bf16(qf[ks], b1, S1b);
      S2a = wmma_bf16(vf[ks], b0, S2a);
      S2b = wmma_bf16(vf[ks], b1, S2b);
    }
    online_update(S1a, S1b, m1, s1, O1, &pbuf[wid][0][0]);
    online_update(S2a, S2b, m2, s2, O2, &pbuf[wid][1][0]);
    v16bf p1 = load_a_frag(&pbuf[wid][0][0], 32);
    v16bf p2 = load_a_frag(&pbuf[wid][1][0], 32);
#pragma unroll
    for (int j = 0; j < 8; ++j) {  // O[16x128] += P[16x32] @ Kblk[32x128]
      v16bf bb = load_b_frag(k2b + (j * 16) * 32, 32);
      O1[j] = wmma_bf16(p1, bb, O1[j]);
      O2[j] = wmma_bf16(p2, bb, O2[j]);
    }
  }

  // H[b][t][head*128 + f] = 0.5*O1/s1 + 0.5*O2/s2
  __bf16* Hrow = Hb + (size_t)(bh >> 3) * L_ * D_ + (size_t)m0 * D_ + (size_t)(bh & 7) * DK_;
#pragma unroll
  for (int j = 0; j < 8; ++j) {
#pragma unroll
    for (int v = 0; v < 8; ++v) {
      float val = O1[j][v] * (0.5f / s1[v]) + O2[j][v] * (0.5f / s2[v]);
      Hrow[(size_t)(v + 8 * half) * D_ + (j * 16 + ln)] = (__bf16)val;
    }
  }
}

// ---------------- Kernel 4: out = H @ Wu^T + bu ----------------
__global__ void ia_out(const __bf16* __restrict__ Hb, const __bf16* __restrict__ Wu,
                       const float* __restrict__ bu, float* __restrict__ out) {
  int w = (blockIdx.x * blockDim.x + threadIdx.x) >> 5;
  int t0 = (w >> 5) * 32;
  int n0 = (w & 31) * 32;
  v8f acc[2][2] = {};
#pragma unroll 4
  for (int kk = 0; kk < D_; kk += 32) {
    v16bf a0 = load_a_frag(Hb + (size_t)t0 * D_ + kk, D_);
    v16bf a1 = load_a_frag(Hb + (size_t)(t0 + 16) * D_ + kk, D_);
    v16bf b0 = load_b_frag(Wu + (size_t)n0 * D_ + kk, D_);
    v16bf b1 = load_b_frag(Wu + (size_t)(n0 + 16) * D_ + kk, D_);
    acc[0][0] = wmma_bf16(a0, b0, acc[0][0]);
    acc[0][1] = wmma_bf16(a0, b1, acc[0][1]);
    acc[1][0] = wmma_bf16(a1, b0, acc[1][0]);
    acc[1][1] = wmma_bf16(a1, b1, acc[1][1]);
  }
  const int lane = threadIdx.x & 31, half = lane >> 4, nn = lane & 15;
#pragma unroll
  for (int j = 0; j < 2; ++j) {
    float bias = bu[n0 + j * 16 + nn];
#pragma unroll
    for (int i = 0; i < 2; ++i)
#pragma unroll
      for (int v = 0; v < 8; ++v)
        out[(size_t)(t0 + i * 16 + v + 8 * half) * D_ + (n0 + j * 16 + nn)] =
            acc[i][j][v] + bias;
  }
}

// ---------------- launch ----------------
extern "C" void kernel_launch(void* const* d_in, const int* in_sizes, int n_in,
                              void* d_out, int out_size, void* d_ws, size_t ws_size,
                              hipStream_t stream) {
  const float* keys    = (const float*)d_in[0];
  const float* queries = (const float*)d_in[1];
  const float* values  = (const float*)d_in[2];
  const float* Wk      = (const float*)d_in[3];
  const float* Wq      = (const float*)d_in[4];
  const float* Wv      = (const float*)d_in[5];
  const float* Wu      = (const float*)d_in[6];
  const float* bu      = (const float*)d_in[7];

  const size_t XN = (size_t)B_ * L_ * D_;  // 8388608
  const size_t WN = (size_t)D_ * D_;       // 1048576

  char* p = (char*)d_ws;
  __bf16* Xk = (__bf16*)p; p += XN * 2;
  __bf16* Xq = (__bf16*)p; p += XN * 2;
  __bf16* Xv = (__bf16*)p; p += XN * 2;
  __bf16* Wkb = (__bf16*)p; p += WN * 2;
  __bf16* Wqb = (__bf16*)p; p += WN * 2;
  __bf16* Wvb = (__bf16*)p; p += WN * 2;
  __bf16* Wub = (__bf16*)p; p += WN * 2;
  __bf16* Qt  = (__bf16*)p; p += XN * 2;
  __bf16* Kt  = (__bf16*)p; p += XN * 2;
  __bf16* Vt  = (__bf16*)p; p += XN * 2;
  __bf16* K2  = (__bf16*)p; p += XN * 2;
  __bf16* Hb  = (__bf16*)p; p += XN * 2;

  ia_cvt_bf16<<<(int)(XN / 4 / 256), 256, 0, stream>>>(keys,    Xk, (int)(XN / 4));
  ia_cvt_bf16<<<(int)(XN / 4 / 256), 256, 0, stream>>>(queries, Xq, (int)(XN / 4));
  ia_cvt_bf16<<<(int)(XN / 4 / 256), 256, 0, stream>>>(values,  Xv, (int)(XN / 4));
  ia_cvt_bf16<<<(int)(WN / 4 / 256), 256, 0, stream>>>(Wk, Wkb, (int)(WN / 4));
  ia_cvt_bf16<<<(int)(WN / 4 / 256), 256, 0, stream>>>(Wq, Wqb, (int)(WN / 4));
  ia_cvt_bf16<<<(int)(WN / 4 / 256), 256, 0, stream>>>(Wv, Wvb, (int)(WN / 4));
  ia_cvt_bf16<<<(int)(WN / 4 / 256), 256, 0, stream>>>(Wu, Wub, (int)(WN / 4));

  ia_proj<<<1024, 256, 0, stream>>>(Xq, Wqb, Qt);
  ia_proj<<<1024, 256, 0, stream>>>(Xk, Wkb, Kt);
  ia_proj<<<1024, 256, 0, stream>>>(Xv, Wvb, Vt);

  ia_transpose<<<dim3(DK_ / 32, L_ / 32, BH_), 256, 0, stream>>>(Kt, K2);

  ia_attn<<<512, 256, 0, stream>>>(Qt, Kt, Vt, K2, Hb);

  ia_out<<<1024, 256, 0, stream>>>(Hb, Wub, bu, (float*)d_out);
}
